// GBEncoder_36575941492864
// MI455X (gfx1250) — matrix operand order repaired and checked by reading, not compiled, not run on hardware
//
#include <hip/hip_runtime.h>
#include <hip/hip_bf16.h>

// ---------------------------------------------------------------------------
// GBEncoder forward for MI455X (gfx1250, wave32, WMMA).
// B=2, N=8192. Heavy GEMMs (pairwise-distance H*H^T and 1x1 convs) run on
// v_wmma_f32_16x16x32_bf16; distance matrices are tiled through LDS and fused
// with top-k selection so they never touch HBM.  Weights are pre-swizzled into
// fragment-major bf16 layout so every lane's B operand is two b128 loads.
// ---------------------------------------------------------------------------

#define BATCH 2
#define NPTS  8192
#define MTOT  (BATCH * NPTS)   // 16384 points total
#define EPSBN 1e-5f

typedef __attribute__((ext_vector_type(16))) __bf16 v16bf;
typedef __attribute__((ext_vector_type(8)))  float  v8f;

union ABu { v16bf v; unsigned short u[16]; };

__device__ __forceinline__ unsigned short f2bf(float f) {
    unsigned int u = __float_as_uint(f);
    u += 0x7FFFu + ((u >> 16) & 1u);          // round-to-nearest-even
    return (unsigned short)(u >> 16);
}
__device__ __forceinline__ float bf2f(unsigned short h) {
    return __uint_as_float(((unsigned int)h) << 16);
}

// ---------------------------------------------------------------------------
// Kernel 1: knn(k=32) on xyz + local 3x3 covariance -> 12-ch features (f32).
// One thread per point; exact f32 like the reference.
// ---------------------------------------------------------------------------
__global__ void knn_cov_kernel(const float* __restrict__ x, float* __restrict__ f12) {
    int p = blockIdx.x * blockDim.x + threadIdx.x;
    if (p >= MTOT) return;
    int b = p / NPTS, base = b * NPTS;
    float xi = x[3 * (size_t)p], yi = x[3 * (size_t)p + 1], zi = x[3 * (size_t)p + 2];
    float sqi = xi * xi + yi * yi + zi * zi;

    float nd[32]; int ni[32];
    for (int t = 0; t < 32; t++) { nd[t] = 3.0e38f; ni[t] = 0; }
    float cmax = 3.0e38f; int cslot = 0;

    for (int j = 0; j < NPTS; j++) {
        size_t q = 3 * (size_t)(base + j);
        float ax = x[q], ay = x[q + 1], az = x[q + 2];
        float d = sqi + ax * ax + ay * ay + az * az
                  - 2.0f * (xi * ax + yi * ay + zi * az);
        if (d < cmax) {
            nd[cslot] = d; ni[cslot] = j;
            cmax = -3.0e38f;
            for (int t = 0; t < 32; t++)
                if (nd[t] > cmax) { cmax = nd[t]; cslot = t; }
        }
    }
    float mx = 0.f, my = 0.f, mz = 0.f;
    for (int t = 0; t < 32; t++) {
        size_t q = 3 * (size_t)(base + ni[t]);
        mx += x[q]; my += x[q + 1]; mz += x[q + 2];
    }
    mx *= (1.f / 32.f); my *= (1.f / 32.f); mz *= (1.f / 32.f);
    float cxx = 0, cxy = 0, cxz = 0, cyy = 0, cyz = 0, czz = 0;
    for (int t = 0; t < 32; t++) {
        size_t q = 3 * (size_t)(base + ni[t]);
        float dx = x[q] - mx, dy = x[q + 1] - my, dz = x[q + 2] - mz;
        cxx += dx * dx; cxy += dx * dy; cxz += dx * dz;
        cyy += dy * dy; cyz += dy * dz; czz += dz * dz;
    }
    const float s = 1.f / 32.f;
    float* o = f12 + (size_t)p * 12;
    o[0] = xi; o[1] = yi; o[2] = zi;
    o[3] = cxx * s; o[4]  = cxy * s; o[5]  = cxz * s;
    o[6] = cxy * s; o[7]  = cyy * s; o[8]  = cyz * s;
    o[9] = cxz * s; o[10] = cyz * s; o[11] = czz * s;
}

// ---------------------------------------------------------------------------
// f32 -> bf16 converters.
// pad_x_bf16: row-major activations, zero-pad trailing columns.
// pad_w_frag: weights pre-swizzled into WMMA-B fragment-major layout:
//   dst[(((ct*KC + kc)*32 + lane)*16 + v]  where
//   lane<16 : col = ct*16+lane,    k = kc*32 + v
//   lane>=16: col = ct*16+lane-16, k = kc*32 + 16 + v
// so each lane's 16 B values are contiguous (two b128 loads).
// ---------------------------------------------------------------------------
__global__ void pad_x_bf16(const float* __restrict__ src, unsigned short* __restrict__ dst,
                           int Csrc, int Cdst) {
    int i = blockIdx.x * blockDim.x + threadIdx.x;
    int total = MTOT * Cdst;
    if (i >= total) return;
    int r = i / Cdst, c = i % Cdst;
    dst[i] = (c < Csrc) ? f2bf(src[(size_t)r * Csrc + c]) : (unsigned short)0;
}

__global__ void pad_w_frag(const float* __restrict__ src, unsigned short* __restrict__ dst,
                           int Ksrc, int Kdst, int Nout) {
    int i = blockIdx.x * blockDim.x + threadIdx.x;
    int KC = Kdst >> 5;
    int total = (Nout >> 4) * KC * 512;
    if (i >= total) return;
    int v     = i & 15;
    int lane  = (i >> 4) & 31;
    int chunk = i >> 9;            // ct*KC + kc
    int kc    = chunk % KC;
    int ct    = chunk / KC;
    int l16   = lane & 15, half = lane >> 4;
    int col   = ct * 16 + l16;
    int k     = kc * 32 + half * 16 + v;
    dst[i] = (k < Ksrc) ? f2bf(src[(size_t)k * Nout + col]) : (unsigned short)0;
}

// Row squared-norms from the bf16-rounded features (keeps dist(i,i) ~ 0,
// consistent with the bf16 WMMA dot products).
__global__ void row_sqnorm(const unsigned short* __restrict__ hb, int C,
                           float* __restrict__ sq) {
    int p = blockIdx.x * blockDim.x + threadIdx.x;
    if (p >= MTOT) return;
    const unsigned short* r = hb + (size_t)p * C;
    float s = 0.f;
    for (int c = 0; c < C; c++) { float f = bf2f(r[c]); s += f * f; }
    sq[p] = s;
}

// ---------------------------------------------------------------------------
// WMMA GEMM: Y[MTOT,Nout] = Xb[MTOT,K] @ W(frag-swizzled)[K,Nout], f32 out.
// One wave per 16x16 tile; block = 4 waves covering 64 output columns.
// A: two contiguous 16B loads/lane; B: two contiguous 16B loads/lane.
// ---------------------------------------------------------------------------
__global__ void wmma_gemm_bf16(const unsigned short* __restrict__ Xb, int K,
                               const unsigned short* __restrict__ Wf, int Nout,
                               float* __restrict__ Y) {
    const int lane = threadIdx.x & 31, wave = threadIdx.x >> 5;
    const int l16 = lane & 15, half = lane >> 4;
    const int r0 = blockIdx.x * 16;
    const int c0 = (blockIdx.y * 4 + wave) * 16;
    const int KC = K >> 5;
    v8f acc = {};
    const int kb0 = half * 16, kb1 = half * 16 + 8;
    const unsigned short* Arow  = Xb + (size_t)(r0 + l16) * K;
    const unsigned short* Bbase = Wf + (((size_t)(c0 >> 4) * KC) << 9) + lane * 16;
    for (int kc = 0; kc < KC; kc++) {
        ABu a, bfr;
        const unsigned short* Ap = Arow + (kc << 5);
#pragma unroll
        for (int i = 0; i < 8; i++) a.u[i]     = Ap[kb0 + i];
#pragma unroll
        for (int i = 0; i < 8; i++) a.u[8 + i] = Ap[kb1 + i];
        const unsigned short* Bp = Bbase + ((size_t)kc << 9);
#pragma unroll
        for (int i = 0; i < 16; i++) bfr.u[i] = Bp[i];
        acc = __builtin_amdgcn_wmma_f32_16x16x32_bf16(false, a.v, false, bfr.v,
                                                      (short)0, acc, false, false);
    }
    float* out = Y + (size_t)(r0 + half * 8) * Nout + c0 + l16;
#pragma unroll
    for (int r = 0; r < 8; r++) out[(size_t)r * Nout] = acc[r];
}

// ---------------------------------------------------------------------------
// BatchNorm (training mode, biased var over all 16384 rows) + activation.
// ---------------------------------------------------------------------------
__global__ void bn_stats(const float* __restrict__ Y, int C,
                         float* __restrict__ mean, float* __restrict__ var) {
    __shared__ float s1[256], s2[256];
    int c = blockIdx.x, t = threadIdx.x;
    float a = 0.f, b = 0.f;
    for (int i = t; i < MTOT; i += 256) {
        float v = Y[(size_t)i * C + c];
        a += v; b += v * v;
    }
    s1[t] = a; s2[t] = b; __syncthreads();
    for (int o = 128; o > 0; o >>= 1) {
        if (t < o) { s1[t] += s1[t + o]; s2[t] += s2[t + o]; }
        __syncthreads();
    }
    if (t == 0) {
        float m = s1[0] / (float)MTOT;
        mean[c] = m;
        var[c]  = s2[0] / (float)MTOT - m * m;
    }
}

__global__ void bn_apply(float* __restrict__ Y, int total, int C,
                         const float* __restrict__ mean, const float* __restrict__ var,
                         const float* __restrict__ g, const float* __restrict__ bt,
                         int leaky) {
    int i = blockIdx.x * blockDim.x + threadIdx.x;
    if (i >= total) return;
    int c = i % C;
    float v = (Y[i] - mean[c]) * rsqrtf(var[c] + EPSBN) * g[c] + bt[c];
    Y[i] = leaky ? (v > 0.f ? v : 0.01f * v) : fmaxf(v, 0.f);
}

// ---------------------------------------------------------------------------
// Fused knn(k=8) + neighbor-feature max.  Block = 128 threads (4 waves) owns
// 16 rows of one batch.  Per 64-column chunk: each wave computes one 16x16
// distance tile via WMMA (dist = |a|^2 + |b|^2 - 2 a.b), tiles staged in LDS.
// Selection is fully parallel: 8 threads per row each keep a private top-8 of
// their 1/8 column subset; one 64-candidate merge per row at the end.
// ---------------------------------------------------------------------------
template <int C>
__global__ void knn_max_kernel(const unsigned short* __restrict__ hbf,
                               const float* __restrict__ sq,
                               const float* __restrict__ hf,
                               float* __restrict__ out) {
    __shared__ float sdist[16][68];
    __shared__ int   smi[16][64];
    __shared__ int   sIdx[16][8];
    const int tid = threadIdx.x;
    const int lane = tid & 31, wave = tid >> 5;
    const int l16 = lane & 15, half = lane >> 4;
    const int nb = NPTS / 16;
    const int b  = blockIdx.x / nb;
    const int rb = blockIdx.x % nb;
    const int r0 = b * NPTS + rb * 16;          // global row base
    const int selRow = tid >> 3, selSub = tid & 7;

    // A fragments for this block's 16 rows (all K chunks), kept in VGPRs.
    ABu afr[C / 32];
    const unsigned short* Arow = hbf + (size_t)(r0 + l16) * C;
    const int kb0 = half * 16, kb1 = half * 16 + 8;
#pragma unroll
    for (int kc = 0; kc < C / 32; kc++) {
#pragma unroll
        for (int i = 0; i < 8; i++) afr[kc].u[i]     = Arow[kc * 32 + kb0 + i];
#pragma unroll
        for (int i = 0; i < 8; i++) afr[kc].u[8 + i] = Arow[kc * 32 + kb1 + i];
    }
    float sqr[8];
#pragma unroll
    for (int r = 0; r < 8; r++) sqr[r] = sq[r0 + half * 8 + r];

    // Private top-8 of this thread's column subset (all 128 threads active).
    float nd8[8]; int ni8[8];
#pragma unroll
    for (int t = 0; t < 8; t++) { nd8[t] = 3.0e38f; ni8[t] = 0; }
    float cmax = 3.0e38f; int cslot = 0;

    for (int cc0 = 0; cc0 < NPTS; cc0 += 64) {
        const int ct0 = b * NPTS + cc0 + wave * 16;     // this wave's col tile
        const float sqc = sq[ct0 + l16];
        v8f acc = {};
        const unsigned short* Bp0 = hbf + (size_t)(ct0 + l16) * C;
#pragma unroll
        for (int kc = 0; kc < C / 32; kc++) {
            ABu bfr;
#pragma unroll
            for (int i = 0; i < 16; i++) bfr.u[i] = Bp0[kc * 32 + half * 16 + i];
            acc = __builtin_amdgcn_wmma_f32_16x16x32_bf16(false, afr[kc].v, false, bfr.v,
                                                          (short)0, acc, false, false);
        }
#pragma unroll
        for (int r = 0; r < 8; r++) {
            float d = sqr[r] + sqc - 2.0f * acc[r];
            sdist[r + half * 8][wave * 16 + l16] = d;
        }
        __syncthreads();
#pragma unroll
        for (int q = 0; q < 8; q++) {
            int cc = selSub * 8 + q;
            float d = sdist[selRow][cc];
            if (d < cmax) {
                nd8[cslot] = d; ni8[cslot] = cc0 + cc;
                cmax = -3.0e38f;
#pragma unroll
                for (int t = 0; t < 8; t++)
                    if (nd8[t] > cmax) { cmax = nd8[t]; cslot = t; }
            }
        }
        __syncthreads();
    }

    // Merge: each row's 8 private top-8 lists -> final top-8.
#pragma unroll
    for (int t = 0; t < 8; t++) {
        sdist[selRow][selSub * 8 + t] = nd8[t];
        smi[selRow][selSub * 8 + t]   = ni8[t];
    }
    __syncthreads();
    if (tid < 16) {
        float md8[8]; int mi8[8];
#pragma unroll
        for (int t = 0; t < 8; t++) { md8[t] = 3.0e38f; mi8[t] = 0; }
        float mmax = 3.0e38f; int mslot = 0;
        for (int cc = 0; cc < 64; cc++) {
            float d = sdist[tid][cc];
            if (d < mmax) {
                md8[mslot] = d; mi8[mslot] = smi[tid][cc];
                mmax = -3.0e38f;
#pragma unroll
                for (int t = 0; t < 8; t++)
                    if (md8[t] > mmax) { mmax = md8[t]; mslot = t; }
            }
        }
#pragma unroll
        for (int t = 0; t < 8; t++) sIdx[tid][t] = mi8[t];
    }
    __syncthreads();

    // Gather elementwise max of the 8 nearest neighbors' f32 features.
    for (int e = tid; e < 16 * C; e += 128) {
        int rr = e / C, c = e % C;
        float m = -3.0e38f;
#pragma unroll
        for (int t = 0; t < 8; t++) {
            int j = sIdx[rr][t];
            m = fmaxf(m, hf[(size_t)(b * NPTS + j) * C + c]);
        }
        out[(size_t)(r0 + rr) * C + c] = m;
    }
}

// ---------------------------------------------------------------------------
// Global max-pool over the 8192 points of each batch (1024 channels).
// ---------------------------------------------------------------------------
__global__ void maxpool_kernel(const float* __restrict__ Y, float* __restrict__ pooled) {
    int i = blockIdx.x * blockDim.x + threadIdx.x;   // 0..2047
    if (i >= BATCH * 1024) return;
    int b = i >> 10, c = i & 1023;
    const float* p = Y + (size_t)b * NPTS * 1024 + c;
    float m = -3.0e38f;
    for (int n = 0; n < NPTS; n++) m = fmaxf(m, p[(size_t)n * 1024]);
    pooled[i] = m;
}

// ---------------------------------------------------------------------------
// Final fused layer: [2,1024] @ W4[1024,512], BN over the 2 samples, ReLU.
// Per-channel independence -> no cross-thread communication needed.
// ---------------------------------------------------------------------------
__global__ void final_kernel(const float* __restrict__ pooled,
                             const float* __restrict__ W4,
                             const float* __restrict__ g4,
                             const float* __restrict__ b4,
                             float* __restrict__ out) {
    int c = threadIdx.x;                              // 0..511
    float y0 = 0.f, y1 = 0.f;
    for (int k = 0; k < 1024; k++) {
        float w = W4[(size_t)k * 512 + c];
        y0 += pooled[k] * w;
        y1 += pooled[1024 + k] * w;
    }
    float m  = 0.5f * (y0 + y1);
    float v  = 0.5f * ((y0 - m) * (y0 - m) + (y1 - m) * (y1 - m));
    float rs = rsqrtf(v + EPSBN);
    out[c]       = fmaxf((y0 - m) * rs * g4[c] + b4[c], 0.f);
    out[512 + c] = fmaxf((y1 - m) * rs * g4[c] + b4[c], 0.f);
}

// ---------------------------------------------------------------------------
// Workspace layout (~81 MB total).
// ---------------------------------------------------------------------------
static constexpr size_t OFF_F12 = 0;                              // 16384*12*4
static constexpr size_t OFF_Y   = (size_t)1 << 20;                // 64 MB f32
static constexpr size_t OFF_H   = OFF_Y  + ((size_t)64 << 20);    // 8 MB f32
static constexpr size_t OFF_XB  = OFF_H  + ((size_t)8  << 20);    // 4 MB bf16
static constexpr size_t OFF_WB  = OFF_XB + ((size_t)4  << 20);    // bf16 weights
static constexpr size_t OFF_SQ  = OFF_WB + ((size_t)1  << 20);    // 64 KB
static constexpr size_t OFF_ST  = OFF_SQ + ((size_t)1  << 20);    // mean/var
static constexpr size_t OFF_PL  = OFF_ST + ((size_t)1  << 20);    // pooled

extern "C" void kernel_launch(void* const* d_in, const int* in_sizes, int n_in,
                              void* d_out, int out_size, void* d_ws, size_t ws_size,
                              hipStream_t stream) {
    (void)in_sizes; (void)n_in; (void)out_size; (void)ws_size;
    const float* x   = (const float*)d_in[0];
    const float* W1  = (const float*)d_in[1];
    const float* g1  = (const float*)d_in[2];
    const float* b1  = (const float*)d_in[3];
    const float* W2  = (const float*)d_in[4];
    const float* g2  = (const float*)d_in[5];
    const float* b2  = (const float*)d_in[6];
    const float* W3  = (const float*)d_in[7];
    const float* g3  = (const float*)d_in[8];
    const float* b3  = (const float*)d_in[9];
    const float* gW1 = (const float*)d_in[10];
    const float* gg1 = (const float*)d_in[11];
    const float* gb1 = (const float*)d_in[12];
    const float* gW2 = (const float*)d_in[13];
    const float* gg2 = (const float*)d_in[14];
    const float* gb2 = (const float*)d_in[15];
    const float* W4  = (const float*)d_in[16];
    const float* g4  = (const float*)d_in[17];
    const float* b4  = (const float*)d_in[18];

    char* w = (char*)d_ws;
    float*          F12  = (float*)(w + OFF_F12);
    float*          Y    = (float*)(w + OFF_Y);
    float*          H    = (float*)(w + OFF_H);
    unsigned short* XB   = (unsigned short*)(w + OFF_XB);
    unsigned short* WB   = (unsigned short*)(w + OFF_WB);
    unsigned short* wb1  = WB;              // 32*64     = 2048
    unsigned short* wb2  = WB + 2048;       // 64*64     = 4096
    unsigned short* wb3  = WB + 6144;       // 64*64     = 4096
    unsigned short* wg1  = WB + 10240;      // 64*128    = 8192
    unsigned short* wg2  = WB + 18432;      // 128*1024  = 131072
    float*          SQ   = (float*)(w + OFF_SQ);
    float*          MEAN = (float*)(w + OFF_ST);
    float*          VAR  = MEAN + 1024;
    float*          PL   = (float*)(w + OFF_PL);

    const dim3 b256(256);
    auto g1d = [](int total) { return dim3((total + 255) / 256); };

    // 1) knn(k=32) on xyz + local covariance -> 12-ch features
    knn_cov_kernel<<<g1d(MTOT), b256, 0, stream>>>(x, F12);

    // Weight conversions: bf16, fragment-major swizzle, K padded to mult of 32
    pad_w_frag<<<g1d(32 * 64),    b256, 0, stream>>>(W1,  wb1, 12,  32,  64);
    pad_w_frag<<<g1d(64 * 64),    b256, 0, stream>>>(W2,  wb2, 64,  64,  64);
    pad_w_frag<<<g1d(64 * 64),    b256, 0, stream>>>(W3,  wb3, 64,  64,  64);
    pad_w_frag<<<g1d(64 * 128),   b256, 0, stream>>>(gW1, wg1, 64,  64,  128);
    pad_w_frag<<<g1d(128 * 1024), b256, 0, stream>>>(gW2, wg2, 128, 128, 1024);

    // 2) h1 = relu(bn(feat12 @ W1))      [MTOT,64]
    pad_x_bf16<<<g1d(MTOT * 32), b256, 0, stream>>>(F12, XB, 12, 32);
    wmma_gemm_bf16<<<dim3(MTOT / 16, 1), dim3(128), 0, stream>>>(XB, 32, wb1, 64, Y);
    bn_stats<<<dim3(64), b256, 0, stream>>>(Y, 64, MEAN, VAR);
    bn_apply<<<g1d(MTOT * 64), b256, 0, stream>>>(Y, MTOT * 64, 64, MEAN, VAR, g1, b1, 0);

    // 3) h2 = relu(bn(h1 @ W2))          [MTOT,64]
    pad_x_bf16<<<g1d(MTOT * 64), b256, 0, stream>>>(Y, XB, 64, 64);
    wmma_gemm_bf16<<<dim3(MTOT / 16, 1), dim3(128), 0, stream>>>(XB, 64, wb2, 64, H);
    bn_stats<<<dim3(64), b256, 0, stream>>>(H, 64, MEAN, VAR);
    bn_apply<<<g1d(MTOT * 64), b256, 0, stream>>>(H, MTOT * 64, 64, MEAN, VAR, g2, b2, 0);

    // 4) h3 = relu(bn(h2 @ W3))          [MTOT,64]
    pad_x_bf16<<<g1d(MTOT * 64), b256, 0, stream>>>(H, XB, 64, 64);
    wmma_gemm_bf16<<<dim3(MTOT / 16, 1), dim3(128), 0, stream>>>(XB, 64, wb3, 64, Y);
    bn_stats<<<dim3(64), b256, 0, stream>>>(Y, 64, MEAN, VAR);
    bn_apply<<<g1d(MTOT * 64), b256, 0, stream>>>(Y, MTOT * 64, 64, MEAN, VAR, g3, b3, 0);

    // 5) n1 = max over knn(h3, k=8)      [MTOT,64]   (WMMA distances, fused top-8)
    pad_x_bf16<<<g1d(MTOT * 64), b256, 0, stream>>>(Y, XB, 64, 64);
    row_sqnorm<<<g1d(MTOT), b256, 0, stream>>>(XB, 64, SQ);
    knn_max_kernel<64><<<dim3(BATCH * NPTS / 16), dim3(128), 0, stream>>>(XB, SQ, Y, H);

    // 6) h4 = leaky(bn(n1 @ gW1))        [MTOT,128]
    pad_x_bf16<<<g1d(MTOT * 64), b256, 0, stream>>>(H, XB, 64, 64);
    wmma_gemm_bf16<<<dim3(MTOT / 16, 2), dim3(128), 0, stream>>>(XB, 64, wg1, 128, Y);
    bn_stats<<<dim3(128), b256, 0, stream>>>(Y, 128, MEAN, VAR);
    bn_apply<<<g1d(MTOT * 128), b256, 0, stream>>>(Y, MTOT * 128, 128, MEAN, VAR, gg1, gb1, 1);

    // 7) n2 = max over knn(h4, k=8)      [MTOT,128]
    pad_x_bf16<<<g1d(MTOT * 128), b256, 0, stream>>>(Y, XB, 128, 128);
    row_sqnorm<<<g1d(MTOT), b256, 0, stream>>>(XB, 128, SQ);
    knn_max_kernel<128><<<dim3(BATCH * NPTS / 16), dim3(128), 0, stream>>>(XB, SQ, Y, H);

    // 8) h5 = leaky(bn(n2 @ gW2))        [MTOT,1024]
    pad_x_bf16<<<g1d(MTOT * 128), b256, 0, stream>>>(H, XB, 128, 128);
    wmma_gemm_bf16<<<dim3(MTOT / 16, 16), dim3(128), 0, stream>>>(XB, 128, wg2, 1024, Y);
    bn_stats<<<dim3(1024), b256, 0, stream>>>(Y, 1024, MEAN, VAR);
    bn_apply<<<g1d(MTOT * 1024), b256, 0, stream>>>(Y, MTOT * 1024, 1024, MEAN, VAR, gg2, gb2, 1);

    // 9) global max-pool + final 1024->512 with BN(B=2) + ReLU
    maxpool_kernel<<<g1d(BATCH * 1024), b256, 0, stream>>>(Y, PL);
    final_kernel<<<dim3(1), dim3(512), 0, stream>>>(PL, W4, g4, b4, (float*)d_out);
}
